// RQVAEAutoencoder_70325794505348
// MI455X (gfx1250) — matrix operand order repaired and checked by reading, not compiled, not run on hardware
//
#include <hip/hip_runtime.h>

// RQ-VAE forward for MI455X (gfx1250, wave32).
// Compute-bound (~160 GFLOP vs ~170MB data) -> f16 WMMA (v_wmma_f32_16x16x32_f16)
// with fp32 accumulation; fp32 kept for argmin/residual/loss math.
// GEMM: 128x128 block tile, 8 waves, 32x64 wave tile (8 accumulators),
// register double-buffered K loop (K % 64 == 0 for all layers here).

typedef _Float16 f16;
typedef __attribute__((ext_vector_type(8)))  _Float16 v8h;
typedef __attribute__((ext_vector_type(16))) _Float16 v16h;
typedef __attribute__((ext_vector_type(8)))  float    v8f;

#define B_ROWS 8192
#define DIM_IN 2048
#define H0V    1024
#define H1V    512
#define DLAT   256
#define QLV    8
#define KCB    2048

// ---------------- elementwise f32 -> f16 ----------------
__global__ void k_cvt(const float* __restrict__ src, f16* __restrict__ dst, int n) {
  int i = blockIdx.x * blockDim.x + threadIdx.x;
  if (i < n) dst[i] = (f16)src[i];
}

// transpose+convert: src (K x N) f32 row-major -> dst (N x K) f16 row-major
__global__ void k_tcvt(const float* __restrict__ src, f16* __restrict__ dst, int K, int N) {
  int i = blockIdx.x * blockDim.x + threadIdx.x;
  if (i < K * N) {
    int k = i / N, n = i - k * N;
    dst[(size_t)n * K + k] = (f16)src[i];
  }
}

// per-codeword squared norms (fp32)
__global__ void k_cnorm(const float* __restrict__ cb, float* __restrict__ cn, int rows, int d) {
  int i = blockIdx.x * blockDim.x + threadIdx.x;
  if (i < rows) {
    const float* p = cb + (size_t)i * d;
    float s = 0.f;
    for (int j = 0; j < d; ++j) s += p[j] * p[j];
    cn[i] = s;
  }
}

// zero quantized accumulator + per-level loss accumulators
__global__ void k_zero(float* __restrict__ q, float* __restrict__ lacc, int n) {
  int i = blockIdx.x * blockDim.x + threadIdx.x;
  if (i < n) q[i] = 0.f;
  if (i < QLV) lacc[i] = 0.f;
}

// ---------------- WMMA fragment loads ----------------
// A 16x32 f16 fragment: lanes0-15 K 0..15, lanes16-31 K 16..31.
// base must already include + hi*16; reads 16 contiguous halfs.
__device__ __forceinline__ v16h ldfragA(const f16* base) {
  v8h lo = *(const v8h*)(base);
  v8h hi8 = *(const v8h*)(base + 8);
  return __builtin_shufflevector(lo, hi8, 0,1,2,3,4,5,6,7,8,9,10,11,12,13,14,15);
}
// B 32x16 fragment: lane = column n; lanes0-15 K 0..15, lanes16-31 K 16..31.
// base must already include + hi*16; reads 16 contiguous halfs.
__device__ __forceinline__ v16h ldfragB(const f16* base) {
  v8h lo = *(const v8h*)(base);
  v8h hi8 = *(const v8h*)(base + 8);
  return __builtin_shufflevector(lo, hi8, 0,1,2,3,4,5,6,7,8,9,10,11,12,13,14,15);
}

// ---------------- WMMA GEMM ----------------
// C[M,N] = op(A[M,K] @ BT[N,K]^T + bias), A/BT f16 row-major, accum f32.
__global__ __launch_bounds__(256)
void k_gemm(const f16* __restrict__ A, const f16* __restrict__ BT,
            const float* __restrict__ bias,
            float* __restrict__ outF, f16* __restrict__ outH,
            int M, int N, int K, int relu) {
  const int tid  = threadIdx.x;
  const int wid  = tid >> 5;
  const int lane = tid & 31;
  const int hl   = lane & 15;   // half-lane: output column / A row within tile
  const int hi   = lane >> 4;   // lane-half selector
  const int wm   = wid & 3;     // 4 row-waves
  const int wn   = wid >> 2;    // 2 col-waves
  const int row0 = blockIdx.x * 128 + wm * 32;
  const int col0 = blockIdx.y * 128 + wn * 64;

  v8f acc[8];
#pragma unroll
  for (int t = 0; t < 8; ++t) acc[t] = (v8f){0.f,0.f,0.f,0.f,0.f,0.f,0.f,0.f};

  // Fragment base pointers (include lane-half offsets).
  const f16* A0 = A + (size_t)(row0 + hl) * K + hi * 16;
  const f16* A1 = A0 + (size_t)16 * K;
  const f16* Bc0 = BT + (size_t)(col0 +  0 + hl) * K + hi * 16;
  const f16* Bc1 = BT + (size_t)(col0 + 16 + hl) * K + hi * 16;
  const f16* Bc2 = BT + (size_t)(col0 + 32 + hl) * K + hi * 16;
  const f16* Bc3 = BT + (size_t)(col0 + 48 + hl) * K + hi * 16;

  v16h a0[2], b0[4], a1[2], b1[4];
  a0[0] = ldfragA(A0);  a0[1] = ldfragA(A1);
  b0[0] = ldfragB(Bc0); b0[1] = ldfragB(Bc1);
  b0[2] = ldfragB(Bc2); b0[3] = ldfragB(Bc3);

  // K is a multiple of 64 for every layer in this workload.
  for (int k0 = 0; k0 < K; k0 += 64) {
    // stage 1: issue loads for k0+32 while WMMAs consume buffer 0
    a1[0] = ldfragA(A0 + k0 + 32);  a1[1] = ldfragA(A1 + k0 + 32);
    b1[0] = ldfragB(Bc0 + k0 + 32); b1[1] = ldfragB(Bc1 + k0 + 32);
    b1[2] = ldfragB(Bc2 + k0 + 32); b1[3] = ldfragB(Bc3 + k0 + 32);
    __builtin_prefetch(A0  + k0 + 96, 0, 3);   // global_prefetch_b8, near scope
    __builtin_prefetch(Bc0 + k0 + 96, 0, 3);
    __builtin_prefetch(Bc1 + k0 + 96, 0, 3);
    __builtin_prefetch(Bc2 + k0 + 96, 0, 3);
    __builtin_prefetch(Bc3 + k0 + 96, 0, 3);
#pragma unroll
    for (int s = 0; s < 2; ++s)
#pragma unroll
      for (int t = 0; t < 4; ++t)
        acc[s * 4 + t] = __builtin_amdgcn_wmma_f32_16x16x32_f16(
            false, a0[s], false, b0[t], (short)0, acc[s * 4 + t], false, false);

    // stage 2: issue loads for k0+64 while WMMAs consume buffer 1
    if (k0 + 64 < K) {
      a0[0] = ldfragA(A0 + k0 + 64);  a0[1] = ldfragA(A1 + k0 + 64);
      b0[0] = ldfragB(Bc0 + k0 + 64); b0[1] = ldfragB(Bc1 + k0 + 64);
      b0[2] = ldfragB(Bc2 + k0 + 64); b0[3] = ldfragB(Bc3 + k0 + 64);
    }
#pragma unroll
    for (int s = 0; s < 2; ++s)
#pragma unroll
      for (int t = 0; t < 4; ++t)
        acc[s * 4 + t] = __builtin_amdgcn_wmma_f32_16x16x32_f16(
            false, a1[s], false, b1[t], (short)0, acc[s * 4 + t], false, false);
  }

  // C/D f32 16x16 layout: lane -> N = hl ; vgpr r -> M = r + hi*8
#pragma unroll
  for (int s = 0; s < 2; ++s) {
#pragma unroll
    for (int t = 0; t < 4; ++t) {
      const int n = col0 + t * 16 + hl;
      const float bv = bias ? bias[n] : 0.f;
#pragma unroll
      for (int r = 0; r < 8; ++r) {
        const int m = row0 + s * 16 + hi * 8 + r;
        float v = acc[s * 4 + t][r] + bv;
        if (relu) v = fmaxf(v, 0.f);
        const size_t o = (size_t)m * N + n;
        if (outF) outF[o] = v;
        if (outH) outH[o] = (f16)v;
      }
    }
  }
}

// ---------------- VQ argmin over 2048 codewords ----------------
__global__ __launch_bounds__(256)
void k_argmin(const float* __restrict__ sc, const float* __restrict__ cn,
              int* __restrict__ idxArr, int q) {
  __shared__ float sv[256];
  __shared__ int   si[256];
  const int b = blockIdx.x, t = threadIdx.x;
  const float* row = sc + (size_t)b * KCB;
  float best = 3.4028235e38f; int bi = 0x7fffffff;
  for (int k = t; k < KCB; k += 256) {
    float d = cn[k] - 2.f * row[k];   // ||r||^2 term is constant per row
    if (d < best) { best = d; bi = k; }
  }
  sv[t] = best; si[t] = bi;
  __syncthreads();
  for (int s = 128; s > 0; s >>= 1) {
    if (t < s) {
      float ov = sv[t + s]; int oi = si[t + s];
      if (ov < sv[t] || (ov == sv[t] && oi < si[t])) { sv[t] = ov; si[t] = oi; }
    }
    __syncthreads();
  }
  if (t == 0) idxArr[b * QLV + q] = si[0];
}

// ---------------- VQ residual/quantized/loss update ----------------
__global__ __launch_bounds__(256)
void k_update(const float* __restrict__ cb, const int* __restrict__ idxArr,
              float* __restrict__ res, f16* __restrict__ resh,
              float* __restrict__ qsum, float* __restrict__ lacc, int q) {
  __shared__ float sv[256];
  const int b = blockIdx.x, t = threadIdx.x;   // blockDim == DLAT == 256
  const int idx = idxArr[b * QLV + q];
  const float c = cb[(size_t)idx * DLAT + t];  // exact fp32 codeword
  const size_t o = (size_t)b * DLAT + t;
  const float r = res[o] - c;
  res[o]  = r;
  resh[o] = (f16)r;
  qsum[o] += c;
  sv[t] = r * r;
  __syncthreads();
  for (int s = 128; s > 0; s >>= 1) {
    if (t < s) sv[t] += sv[t + s];
    __syncthreads();
  }
  if (t == 0) atomicAdd(&lacc[q], sv[0]);
}

// ---------------- emit indices (as float) + losses ----------------
__global__ void k_final(const int* __restrict__ idxArr, const float* __restrict__ lacc,
                        float* __restrict__ out) {
  const int NI = B_ROWS * QLV;
  int i = blockIdx.x * blockDim.x + threadIdx.x;
  if (i < NI) {
    out[(size_t)B_ROWS * DIM_IN + i] = (float)idxArr[i];
  } else if (i < NI + QLV) {
    out[(size_t)B_ROWS * DIM_IN + NI + (i - NI)] =
        lacc[i - NI] / (float)((size_t)B_ROWS * DLAT);
  }
}

extern "C" void kernel_launch(void* const* d_in, const int* in_sizes, int n_in,
                              void* d_out, int out_size, void* d_ws, size_t ws_size,
                              hipStream_t stream) {
  (void)in_sizes; (void)n_in; (void)out_size; (void)ws_size;
  const float* x    = (const float*)d_in[0];
  const float* e_w0 = (const float*)d_in[1];
  const float* e_b0 = (const float*)d_in[2];
  const float* e_w1 = (const float*)d_in[3];
  const float* e_b1 = (const float*)d_in[4];
  const float* e_w2 = (const float*)d_in[5];
  const float* e_b2 = (const float*)d_in[6];
  const float* d_w0 = (const float*)d_in[7];
  const float* d_b0 = (const float*)d_in[8];
  const float* d_w1 = (const float*)d_in[9];
  const float* d_b1 = (const float*)d_in[10];
  const float* d_w2 = (const float*)d_in[11];
  const float* d_b2 = (const float*)d_in[12];
  const float* cbf  = (const float*)d_in[13];
  float* out = (float*)d_out;

  char* ws = (char*)d_ws;
  size_t off = 0;
  auto alloc = [&](size_t bytes) -> void* {
    void* p = ws + off; off += (bytes + 255) & ~(size_t)255; return p;
  };
  f16*   xh   = (f16*)  alloc((size_t)B_ROWS * DIM_IN * 2);
  f16*   w0t  = (f16*)  alloc((size_t)H0V * DIM_IN * 2);
  f16*   w1t  = (f16*)  alloc((size_t)H1V * H0V * 2);
  f16*   w2t  = (f16*)  alloc((size_t)DLAT * H1V * 2);
  f16*   d0t  = (f16*)  alloc((size_t)H1V * DLAT * 2);
  f16*   d1t  = (f16*)  alloc((size_t)H0V * H1V * 2);
  f16*   d2t  = (f16*)  alloc((size_t)DIM_IN * H0V * 2);
  f16*   cbh  = (f16*)  alloc((size_t)QLV * KCB * DLAT * 2);
  float* cn   = (float*)alloc((size_t)QLV * KCB * 4);
  f16*   h0h  = (f16*)  alloc((size_t)B_ROWS * H0V * 2);
  f16*   h1h  = (f16*)  alloc((size_t)B_ROWS * H1V * 2);
  float* res  = (float*)alloc((size_t)B_ROWS * DLAT * 4);
  f16*   resh = (f16*)  alloc((size_t)B_ROWS * DLAT * 2);
  float* qsum = (float*)alloc((size_t)B_ROWS * DLAT * 4);
  f16*   qh   = (f16*)  alloc((size_t)B_ROWS * DLAT * 2);
  float* sc   = (float*)alloc((size_t)B_ROWS * KCB * 4);   // reused per level
  int*   idxA = (int*)  alloc((size_t)B_ROWS * QLV * 4);
  float* lacc = (float*)alloc(256);
  f16*   g0h  = (f16*)  alloc((size_t)B_ROWS * H1V * 2);
  f16*   g1h  = (f16*)  alloc((size_t)B_ROWS * H0V * 2);

  auto nb = [](size_t n) { return (unsigned)((n + 255) / 256); };
  const int TB = 256;

  // --- precision conversion / weight transposes / codebook norms ---
  k_cvt <<<nb((size_t)B_ROWS * DIM_IN), TB, 0, stream>>>(x, xh, B_ROWS * DIM_IN);
  k_tcvt<<<nb((size_t)DIM_IN * H0V), TB, 0, stream>>>(e_w0, w0t, DIM_IN, H0V);
  k_tcvt<<<nb((size_t)H0V * H1V),    TB, 0, stream>>>(e_w1, w1t, H0V, H1V);
  k_tcvt<<<nb((size_t)H1V * DLAT),   TB, 0, stream>>>(e_w2, w2t, H1V, DLAT);
  k_tcvt<<<nb((size_t)DLAT * H1V),   TB, 0, stream>>>(d_w0, d0t, DLAT, H1V);
  k_tcvt<<<nb((size_t)H1V * H0V),    TB, 0, stream>>>(d_w1, d1t, H1V, H0V);
  k_tcvt<<<nb((size_t)H0V * DIM_IN), TB, 0, stream>>>(d_w2, d2t, H0V, DIM_IN);
  k_cvt <<<nb((size_t)QLV * KCB * DLAT), TB, 0, stream>>>(cbf, cbh, QLV * KCB * DLAT);
  k_cnorm<<<nb(QLV * KCB), TB, 0, stream>>>(cbf, cn, QLV * KCB, DLAT);

  // --- encoder ---
  { dim3 g(B_ROWS / 128, H0V / 128);
    k_gemm<<<g, 256, 0, stream>>>(xh,  w0t, e_b0, nullptr, h0h, B_ROWS, H0V, DIM_IN, 1); }
  { dim3 g(B_ROWS / 128, H1V / 128);
    k_gemm<<<g, 256, 0, stream>>>(h0h, w1t, e_b1, nullptr, h1h, B_ROWS, H1V, H0V, 1); }
  { dim3 g(B_ROWS / 128, DLAT / 128);  // z written directly as initial residual (f32 + f16)
    k_gemm<<<g, 256, 0, stream>>>(h1h, w2t, e_b2, res, resh, B_ROWS, DLAT, H1V, 0); }

  k_zero<<<nb((size_t)B_ROWS * DLAT), TB, 0, stream>>>(qsum, lacc, B_ROWS * DLAT);

  // --- residual VQ: 8 levels ---
  { dim3 gs(B_ROWS / 128, KCB / 128);
    for (int q = 0; q < QLV; ++q) {
      k_gemm<<<gs, 256, 0, stream>>>(resh, cbh + (size_t)q * KCB * DLAT, nullptr,
                                     sc, nullptr, B_ROWS, KCB, DLAT, 0);
      k_argmin<<<B_ROWS, 256, 0, stream>>>(sc, cn + (size_t)q * KCB, idxA, q);
      k_update<<<B_ROWS, 256, 0, stream>>>(cbf + (size_t)q * KCB * DLAT, idxA,
                                           res, resh, qsum, lacc, q);
    }
  }

  // --- decoder ---
  k_cvt<<<nb((size_t)B_ROWS * DLAT), TB, 0, stream>>>(qsum, qh, B_ROWS * DLAT);
  { dim3 g(B_ROWS / 128, H1V / 128);
    k_gemm<<<g, 256, 0, stream>>>(qh,  d0t, d_b0, nullptr, g0h, B_ROWS, H1V, DLAT, 1); }
  { dim3 g(B_ROWS / 128, H0V / 128);
    k_gemm<<<g, 256, 0, stream>>>(g0h, d1t, d_b1, nullptr, g1h, B_ROWS, H0V, H1V, 1); }
  { dim3 g(B_ROWS / 128, DIM_IN / 128);
    k_gemm<<<g, 256, 0, stream>>>(g1h, d2t, d_b2, out, nullptr, B_ROWS, DIM_IN, H0V, 0); }

  k_final<<<nb((size_t)B_ROWS * QLV + QLV), TB, 0, stream>>>(idxA, lacc, out);
}
